// MyMultiHeadAttention_14396730376477
// MI455X (gfx1250) — matrix operand order repaired and checked by reading, not compiled, not run on hardware
//
#include <hip/hip_runtime.h>

typedef _Float16 half_t;
typedef __attribute__((ext_vector_type(16))) _Float16 v16h;
typedef __attribute__((ext_vector_type(8)))  _Float16 v8h;
typedef __attribute__((ext_vector_type(4)))  _Float16 v4h;
typedef __attribute__((ext_vector_type(8)))  float    v8f;
typedef __attribute__((ext_vector_type(4)))  float    v4f;

#define NE   1024
#define NL   1024
#define NB   8
#define NH   16
#define NDH  64

#define WMMA16(A,B,C) __builtin_amdgcn_wmma_f32_16x16x32_f16(false,(A),false,(B),(short)0,(C),false,false)

// A fragment (16x32 f16): lane&15 = M row; lanes<16 hold K 0..7 & 16..23, lanes>=16 hold K 8..15 & 24..31
static __device__ __forceinline__ v16h lds_a_frag(const half_t* p0, int stride, int lane) {
  const half_t* p = p0 + (lane & 15) * stride + ((lane >> 4) << 3);
  v8h lo = *(const v8h*)p;
  v8h hi = *(const v8h*)(p + 16);
  v16h r;
#pragma unroll
  for (int i = 0; i < 8; ++i) { r[i] = lo[i]; r[i + 8] = hi[i]; }
  return r;
}

// B fragment (32x16 f16): lane&15 = N col; lanes<16 hold K 0..7 & 16..23, lanes>=16 hold K 8..15 & 24..31
static __device__ __forceinline__ v16h lds_b_frag(const half_t* p0, int stride, int lane) {
  const half_t* p = p0 + (lane & 15) * stride + ((lane >> 4) << 3);
  v8h lo = *(const v8h*)p;
  v8h hi = *(const v8h*)(p + 16);
  v16h r;
#pragma unroll
  for (int i = 0; i < 8; ++i) { r[i] = lo[i]; r[i + 8] = hi[i]; }
  return r;
}

__global__ __launch_bounds__(256) void cvt_f32_f16(const float* __restrict__ in,
                                                   half_t* __restrict__ out, int n4) {
  int i = blockIdx.x * 256 + threadIdx.x;
  if (i < n4) {
    v4f x = *(const v4f*)(in + (size_t)i * 4);
    v4h y;
#pragma unroll
    for (int u = 0; u < 4; ++u) y[u] = (half_t)x[u];
    *(v4h*)(out + (size_t)i * 4) = y;
  }
}

// Q/K/V projection: Out[m,n] = X[m,:] . W[n,:] + bias[n], stored f16 head-split [B,H,L,Dh]
__global__ __launch_bounds__(256) void qkv_gemm(const float* __restrict__ Xq,
                                                const float* __restrict__ Xk,
                                                const float* __restrict__ Xv,
                                                const half_t* __restrict__ Wh,
                                                const float* __restrict__ bpk,
                                                half_t* __restrict__ Qh,
                                                half_t* __restrict__ Kh,
                                                half_t* __restrict__ Vh) {
  const int t = blockIdx.z;
  const float* X = (t == 0) ? Xq : (t == 1) ? Xk : Xv;
  half_t* Out = (t == 0) ? Qh : (t == 1) ? Kh : Vh;
  const half_t* W = Wh + (size_t)t * NE * NE;
  const float* bias = bpk + t * NE;

  const int m0 = blockIdx.x * 128;
  const int n0 = blockIdx.y * 128;

  __shared__ alignas(16) half_t As[128 * 48];
  __shared__ alignas(16) half_t Bs[128 * 48];

  const int tid = threadIdx.x;
  const int lane = tid & 31;
  const int wave = tid >> 5;
  const int wy = wave >> 2;   // 0..1 : 64 M-rows each
  const int wx = wave & 3;    // 0..3 : 32 N-cols each
  const int lane15 = lane & 15;
  const int rbase = (lane >> 4) << 3;

  v8f acc[4][2];
  {
    v8f z = {};
#pragma unroll
    for (int mi = 0; mi < 4; ++mi)
#pragma unroll
      for (int ni = 0; ni < 2; ++ni) acc[mi][ni] = z;
  }

  for (int k0 = 0; k0 < NE; k0 += 32) {
    __syncthreads();
    // A tile 128x32 f32 -> f16 (each thread: 4 x float4)
#pragma unroll
    for (int it = 0; it < 4; ++it) {
      int e = tid * 4 + it * 1024;
      int r = e >> 5, c = e & 31;
      v4f x = *(const v4f*)(X + (size_t)(m0 + r) * NE + k0 + c);
      v4h y;
#pragma unroll
      for (int u = 0; u < 4; ++u) y[u] = (half_t)x[u];
      *(v4h*)(As + r * 48 + c) = y;
    }
    // B tile 128x32 f16 (weight rows are K-contiguous)
#pragma unroll
    for (int it = 0; it < 2; ++it) {
      int e = tid * 8 + it * 2048;
      int r = e >> 5, c = e & 31;
      *(v8h*)(Bs + r * 48 + c) = *(const v8h*)(W + (size_t)(n0 + r) * NE + k0 + c);
    }
    __syncthreads();

    v16h bf[2];
#pragma unroll
    for (int ni = 0; ni < 2; ++ni)
      bf[ni] = lds_b_frag(Bs + (wx * 32 + ni * 16) * 48, 48, lane);
#pragma unroll
    for (int mi = 0; mi < 4; ++mi) {
      v16h a = lds_a_frag(As + (wy * 64 + mi * 16) * 48, 48, lane);
#pragma unroll
      for (int ni = 0; ni < 2; ++ni)
        acc[mi][ni] = WMMA16(a, bf[ni], acc[mi][ni]);
    }
  }

#pragma unroll
  for (int mi = 0; mi < 4; ++mi) {
#pragma unroll
    for (int ni = 0; ni < 2; ++ni) {
      int n = n0 + wx * 32 + ni * 16 + lane15;
      float bn = bias[n];
      int h = n >> 6, d = n & 63;
#pragma unroll
      for (int r = 0; r < 8; ++r) {
        int m = m0 + wy * 64 + mi * 16 + rbase + r;
        int b = m >> 10, l = m & 1023;
        Out[(((size_t)(b * NH + h)) * NL + l) * NDH + d] = (half_t)(acc[mi][ni][r] + bn);
      }
    }
  }
}

// Per (b,h): S = relu(Q K^T / 8) -> attn_weight (f32) ; O = S @ V -> attn_h (f16, [B,L,E])
__global__ __launch_bounds__(256) void attn_relu(const half_t* __restrict__ Qh,
                                                 const half_t* __restrict__ Kh,
                                                 const half_t* __restrict__ Vh,
                                                 float* __restrict__ attnw,
                                                 half_t* __restrict__ attn_h) {
  const int bh = blockIdx.y;        // 0..127
  const int b = bh >> 4, h = bh & 15;
  const int q0 = blockIdx.x * 128;  // 8 tiles of q rows

  const half_t* Q = Qh + (size_t)bh * NL * NDH;
  const half_t* K = Kh + (size_t)bh * NL * NDH;
  const half_t* V = Vh + (size_t)bh * NL * NDH;
  float* Sout = attnw + (size_t)bh * NL * NL;

  __shared__ alignas(16) half_t Qs[128 * 72];
  __shared__ alignas(16) half_t Ks[128 * 72];
  __shared__ alignas(16) half_t Vs[64 * 136];   // transposed: [d][l]
  __shared__ alignas(16) half_t Ss[128 * 136];  // relu'd scores in f16

  const int tid = threadIdx.x;
  const int lane = tid & 31;
  const int wave = tid >> 5;
  const int wy = wave >> 2;   // 0..1 : 64 q-rows
  const int wx = wave & 3;    // 0..3
  const int lane15 = lane & 15;
  const int rbase = (lane >> 4) << 3;

#pragma unroll
  for (int it = 0; it < 4; ++it) {
    int e = tid * 8 + it * 2048;
    int r = e >> 6, c = e & 63;
    *(v8h*)(Qs + r * 72 + c) = *(const v8h*)(Q + (size_t)(q0 + r) * NDH + c);
  }

  v8f oacc[4];
  {
    v8f z = {};
#pragma unroll
    for (int mi = 0; mi < 4; ++mi) oacc[mi] = z;
  }

  for (int j = 0; j < 8; ++j) {
    __syncthreads();
    // Load K tile straight; V tile transposed into [d][l]
#pragma unroll
    for (int it = 0; it < 4; ++it) {
      int e = tid * 8 + it * 2048;
      int r = e >> 6, c = e & 63;
      *(v8h*)(Ks + r * 72 + c) = *(const v8h*)(K + (size_t)(j * 128 + r) * NDH + c);
      v8h x = *(const v8h*)(V + (size_t)(j * 128 + r) * NDH + c);
#pragma unroll
      for (int u = 0; u < 8; ++u) Vs[(c + u) * 136 + r] = x[u];
    }
    __syncthreads();

    // S tile: Q[128x64] @ K^T -> 128x128, each wave 64x32
    v8f s[4][2];
    {
      v8f z = {};
#pragma unroll
      for (int mi = 0; mi < 4; ++mi)
#pragma unroll
        for (int ni = 0; ni < 2; ++ni) s[mi][ni] = z;
    }
#pragma unroll
    for (int kk = 0; kk < 2; ++kk) {
      v16h bf[2];
#pragma unroll
      for (int ni = 0; ni < 2; ++ni)
        bf[ni] = lds_b_frag(Ks + (wx * 32 + ni * 16) * 72 + kk * 32, 72, lane);
#pragma unroll
      for (int mi = 0; mi < 4; ++mi) {
        v16h a = lds_a_frag(Qs + (wy * 64 + mi * 16) * 72 + kk * 32, 72, lane);
#pragma unroll
        for (int ni = 0; ni < 2; ++ni)
          s[mi][ni] = WMMA16(a, bf[ni], s[mi][ni]);
      }
    }

    // scale + relu; emit f32 attn_weight and f16 copy to LDS for the PV GEMM
#pragma unroll
    for (int mi = 0; mi < 4; ++mi) {
#pragma unroll
      for (int ni = 0; ni < 2; ++ni) {
        int n = wx * 32 + ni * 16 + lane15;
#pragma unroll
        for (int r = 0; r < 8; ++r) {
          int m = wy * 64 + mi * 16 + rbase + r;
          float v = s[mi][ni][r] * 0.125f;
          v = v > 0.0f ? v : 0.0f;
          Sout[(size_t)(q0 + m) * NL + j * 128 + n] = v;
          Ss[m * 136 + n] = (half_t)v;
        }
      }
    }
    __syncthreads();

    // O += S_h[128x128] @ V[128x64]; each wave: 64 rows x 16 cols
#pragma unroll
    for (int kk = 0; kk < 4; ++kk) {
      v16h bv = lds_b_frag(Vs + (wx * 16) * 136 + kk * 32, 136, lane);
#pragma unroll
      for (int mi = 0; mi < 4; ++mi) {
        v16h a = lds_a_frag(Ss + (wy * 64 + mi * 16) * 136 + kk * 32, 136, lane);
        oacc[mi] = WMMA16(a, bv, oacc[mi]);
      }
    }
  }

#pragma unroll
  for (int mi = 0; mi < 4; ++mi) {
    int d = wx * 16 + lane15;
#pragma unroll
    for (int r = 0; r < 8; ++r) {
      int m = wy * 64 + mi * 16 + rbase + r;
      attn_h[((size_t)b * NL + q0 + m) * NE + h * NDH + d] = (half_t)oacc[mi][r];
    }
  }
}

// out = attn_h @ W_out^T + b_out (f32 result)
__global__ __launch_bounds__(256) void out_gemm(const half_t* __restrict__ Ah,
                                                const half_t* __restrict__ Wh,
                                                const float* __restrict__ bias,
                                                float* __restrict__ Out) {
  const int m0 = blockIdx.x * 128;
  const int n0 = blockIdx.y * 128;

  __shared__ alignas(16) half_t As[128 * 48];
  __shared__ alignas(16) half_t Bs[128 * 48];

  const int tid = threadIdx.x;
  const int lane = tid & 31;
  const int wave = tid >> 5;
  const int wy = wave >> 2;
  const int wx = wave & 3;
  const int lane15 = lane & 15;
  const int rbase = (lane >> 4) << 3;

  v8f acc[4][2];
  {
    v8f z = {};
#pragma unroll
    for (int mi = 0; mi < 4; ++mi)
#pragma unroll
      for (int ni = 0; ni < 2; ++ni) acc[mi][ni] = z;
  }

  for (int k0 = 0; k0 < NE; k0 += 32) {
    __syncthreads();
#pragma unroll
    for (int it = 0; it < 2; ++it) {
      int e = tid * 8 + it * 2048;
      int r = e >> 5, c = e & 31;
      *(v8h*)(As + r * 48 + c) = *(const v8h*)(Ah + (size_t)(m0 + r) * NE + k0 + c);
      *(v8h*)(Bs + r * 48 + c) = *(const v8h*)(Wh + (size_t)(n0 + r) * NE + k0 + c);
    }
    __syncthreads();

    v16h bf[2];
#pragma unroll
    for (int ni = 0; ni < 2; ++ni)
      bf[ni] = lds_b_frag(Bs + (wx * 32 + ni * 16) * 48, 48, lane);
#pragma unroll
    for (int mi = 0; mi < 4; ++mi) {
      v16h a = lds_a_frag(As + (wy * 64 + mi * 16) * 48, 48, lane);
#pragma unroll
      for (int ni = 0; ni < 2; ++ni)
        acc[mi][ni] = WMMA16(a, bf[ni], acc[mi][ni]);
    }
  }

#pragma unroll
  for (int mi = 0; mi < 4; ++mi) {
#pragma unroll
    for (int ni = 0; ni < 2; ++ni) {
      int n = n0 + wx * 32 + ni * 16 + lane15;
      float bn = bias[n];
#pragma unroll
      for (int r = 0; r < 8; ++r) {
        int m = m0 + wy * 64 + mi * 16 + rbase + r;
        Out[(size_t)m * NE + n] = acc[mi][ni][r] + bn;
      }
    }
  }
}

extern "C" void kernel_launch(void* const* d_in, const int* in_sizes, int n_in,
                              void* d_out, int out_size, void* d_ws, size_t ws_size,
                              hipStream_t stream) {
  const float* query    = (const float*)d_in[0];
  const float* key      = (const float*)d_in[1];
  const float* value    = (const float*)d_in[2];
  const float* W_packed = (const float*)d_in[3];
  const float* b_packed = (const float*)d_in[4];
  const float* W_out    = (const float*)d_in[5];
  const float* b_out    = (const float*)d_in[6];

  // workspace layout (f16): Wqkv[3E*E] | Wout[E*E] | Q | K | V (each [B,H,L,Dh]) | attn [B,L,E]
  half_t* Wqkv_h = (half_t*)d_ws;
  half_t* Wout_h = Wqkv_h + (size_t)3 * NE * NE;
  half_t* Qh     = Wout_h + (size_t)NE * NE;
  half_t* Kh     = Qh + (size_t)NB * NH * NL * NDH;
  half_t* Vh     = Kh + (size_t)NB * NH * NL * NDH;
  half_t* attn_h = Vh + (size_t)NB * NH * NL * NDH;

  float* out   = (float*)d_out;                       // [B, L, E]
  float* attnw = out + (size_t)NB * NL * NE;          // [B, H, L, L]

  cvt_f32_f16<<<(3 * NE * NE / 4 + 255) / 256, 256, 0, stream>>>(W_packed, Wqkv_h, 3 * NE * NE / 4);
  cvt_f32_f16<<<(NE * NE / 4 + 255) / 256, 256, 0, stream>>>(W_out, Wout_h, NE * NE / 4);

  qkv_gemm<<<dim3(64, 8, 3), 256, 0, stream>>>(query, key, value, Wqkv_h, b_packed, Qh, Kh, Vh);
  attn_relu<<<dim3(8, 128), 256, 0, stream>>>(Qh, Kh, Vh, attnw, attn_h);
  out_gemm<<<dim3(64, 8), 256, 0, stream>>>(attn_h, Wout_h, b_out, out);
}
